// Controller_69690139345250
// MI455X (gfx1250) — matrix operand order repaired and checked by reading, not compiled, not run on hardware
//
#include <hip/hip_runtime.h>
#include <hip/hip_bf16.h>

typedef __attribute__((ext_vector_type(16))) _Float16 v16h;
typedef __attribute__((ext_vector_type(8)))  _Float16 v8h;
typedef __attribute__((ext_vector_type(8)))  float    v8f;

#define B_TOTAL   65536
#define ROWS_BLK  64          // 4 waves * 16 batch cols
#define WAVES     4
#define NTILE     25          // 400 gate rows / 16
#define TANH_C    1.5f

// ---- LDS layout ----
// f32 region: gb[400], hb[48]  -> 1792 B
// f16 region (halfs): WIH 12800 | WHH 51200 | WHD 6144 | HT 8192 | XT 2048
#define SMEM_F32_BYTES 1792
#define OFF_WIH 0
#define OFF_WHH 12800
#define OFF_WHD 64000
#define OFF_HT  70144
#define OFF_XT  78336
#define SMEM_BYTES (SMEM_F32_BYTES + (78336 + WAVES*512) * 2)   // 162560

__device__ __forceinline__ float fast_rcp(float x) { return __builtin_amdgcn_rcpf(x); }

#if __has_builtin(__builtin_amdgcn_tanhf)
__device__ __forceinline__ float tanha(float z) { return __builtin_amdgcn_tanhf(z); }
__device__ __forceinline__ float sigf(float z)  {
    return __builtin_fmaf(0.5f, __builtin_amdgcn_tanhf(0.5f * z), 0.5f);
}
#else
__device__ __forceinline__ float tanha(float z) {
    float a = fabsf(z);
    float e = __expf(-2.0f * a);
    float r = (1.0f - e) * fast_rcp(1.0f + e);
    return (z < 0.0f) ? -r : r;
}
__device__ __forceinline__ float sigf(float z) {
    return fast_rcp(1.0f + __expf(-z));
}
#endif

__device__ __forceinline__ v16h load16(const _Float16* p) {
    v8h lo = *(const v8h*)p;
    v8h hi = *(const v8h*)(p + 8);
    return __builtin_shufflevector(lo, hi, 0,1,2,3,4,5,6,7,8,9,10,11,12,13,14,15);
}

__global__ void __launch_bounds__(128)
controller_kernel(const int* __restrict__ policy, const int* __restrict__ label,
                  const int* __restrict__ group_fnc, const float* __restrict__ in_emb,
                  const float* __restrict__ W_ih, const float* __restrict__ W_hh,
                  const float* __restrict__ b_ih, const float* __restrict__ b_hh,
                  const float* __restrict__ o_W, const float* __restrict__ o_b,
                  const float* __restrict__ p_W, const float* __restrict__ p_b,
                  const float* __restrict__ m_W, const float* __restrict__ m_b,
                  const float* __restrict__ o_emb, const float* __restrict__ p_emb,
                  const float* __restrict__ m_emb, float* __restrict__ out)
{
    extern __shared__ char smem[];
    float*    gb  = (float*)smem;           // 400: per (tile,M) gate bias
    float*    hb  = gb + 400;               // 48 : head biases padded to 16
    _Float16* hbase = (_Float16*)(smem + SMEM_F32_BYTES);
    _Float16* WIH = hbase + OFF_WIH;
    _Float16* WHH = hbase + OFF_WHH;
    _Float16* WHD = hbase + OFF_WHD;
    _Float16* HT  = hbase + OFF_HT;
    _Float16* XT  = hbase + OFF_XT;

    const int tid = threadIdx.x;
    const int w   = tid >> 5;       // wave in block
    const int L   = tid & 31;       // lane
    const int n   = L & 15;         // batch column within wave tile
    const int hi  = L >> 4;         // lane-half
    const int blockRow0 = blockIdx.x * ROWS_BLK;

    // ---------------- pack weights into WMMA fragment layouts ----------------
    // A-frag (16x32 f16): lane l: M=l%16; elem j: K = (l/16)*16+j
    // Row permutation: tile t, row M -> gate M%4 of unit u = 4t + 2*(M/8) + (M%8)/4
    for (int idx = tid; idx < 12800; idx += 128) {           // W_ih (K = E = 32)
        int t = idx >> 9, r = idx & 511, lane = r >> 4, j = r & 15;
        int M = lane & 15, hiA = lane >> 4;
        int k = hiA*16 + j;
        int u = t*4 + ((M >> 3) << 1) + ((M & 7) >> 2);
        int grow = (M & 3) * 100 + u;
        WIH[idx] = (_Float16)W_ih[grow*32 + k];
    }
    for (int idx = tid; idx < 51200; idx += 128) {           // W_hh (K = H pad 128)
        int tc = idx >> 9, t = tc >> 2, kc = tc & 3;
        int r = idx & 511, lane = r >> 4, j = r & 15;
        int M = lane & 15, hiA = lane >> 4;
        int kl = hiA*16 + j;
        int kg = kc*32 + kl;
        int u = t*4 + ((M >> 3) << 1) + ((M & 7) >> 2);
        int grow = (M & 3) * 100 + u;
        WHH[idx] = (kg < 100) ? (_Float16)W_hh[grow*100 + kg] : (_Float16)0.0f;
    }
    for (int idx = tid; idx < 6144; idx += 128) {            // head weights (rows = classes)
        int hc = idx >> 9, t = hc >> 2, kc = hc & 3;
        int r = idx & 511, lane = r >> 4, j = r & 15;
        int M = lane & 15, hiA = lane >> 4;
        int kl = hiA*16 + j;
        int kg = kc*32 + kl;
        const float* Wt = (t == 0) ? o_W : ((t == 1) ? p_W : m_W);
        int nhh = (t == 0) ? 15 : ((t == 1) ? 11 : 10);
        WHD[idx] = (M < nhh && kg < 100) ? (_Float16)Wt[M*100 + kg] : (_Float16)0.0f;
    }
    for (int idx = tid; idx < 400; idx += 128) {             // gate bias, fragment order
        int t = idx >> 4, M = idx & 15;
        int u = t*4 + ((M >> 3) << 1) + ((M & 7) >> 2);
        int grow = (M & 3) * 100 + u;
        gb[idx] = b_ih[grow] + b_hh[grow];
    }
    if (tid < 48) {
        int t = tid >> 4, cc = tid & 15;
        const float* bt = (t == 0) ? o_b : ((t == 1) ? p_b : m_b);
        int nhh = (t == 0) ? 15 : ((t == 1) ? 11 : 10);
        hb[tid] = (cc < nhh) ? bt[cc] : 0.0f;
    }
    for (int idx = tid; idx < WAVES*2048; idx += 128) HT[idx] = (_Float16)0.0f;  // h0 = 0
    for (int idx = tid; idx < WAVES*512;  idx += 128) {      // x0 = in_emb[group_fnc[label]]
        int w2 = idx >> 9, r = idx & 511, lane = r >> 4, e = r & 15;
        int n2 = lane & 15, hiB = lane >> 4;
        int k = hiB*16 + e;
        int rowg = blockRow0 + w2*16 + n2;
        int gf = group_fnc[label[rowg]];
        XT[idx] = (_Float16)in_emb[gf*32 + k];
    }
    __syncthreads();

    _Float16* myHT = HT + w * 2048;
    _Float16* myXT = XT + w * 512;

    float cReg[50];
    #pragma unroll
    for (int i = 0; i < 50; ++i) cReg[i] = 0.0f;
    float lpAcc = 0.0f, entAcc = 0.0f;

    int outer = 0, kh = 0;
    for (int s = 0; s < 30; ++s) {
        // ---- B-operand fragments: previous x, previous h (wave-local) ----
        v16h xB = load16(myXT + L*16);
        v16h hB[4];
        #pragma unroll
        for (int kc = 0; kc < 4; ++kc) hB[kc] = load16(myHT + kc*512 + L*16);

        // ---- gates: g^T tiles = W * [x;h]^T, fused elementwise LSTM update ----
        // FULLY UNROLLED so cReg[] indices are compile-time constants (no scratch)
        #pragma unroll
        for (int t = 0; t < NTILE; ++t) {
            const float* bp = gb + t*16 + hi*8;
            v8f acc;
            #pragma unroll
            for (int v = 0; v < 8; ++v) acc[v] = bp[v];
            v16h aIH = load16(WIH + (t*32 + L)*16);
            acc = __builtin_amdgcn_wmma_f32_16x16x32_f16(false, aIH, false, xB,
                                                         (short)0, acc, false, false);
            #pragma unroll
            for (int kc = 0; kc < 4; ++kc) {
                v16h aHH = load16(WHH + ((t*4 + kc)*32 + L)*16);
                acc = __builtin_amdgcn_wmma_f32_16x16x32_f16(false, aHH, false, hB[kc],
                                                             (short)0, acc, false, false);
            }
            // lane holds gates i,f,g,o of units u0,u0+1 for batch col n
            _Float16 hp0, hp1;
            #pragma unroll
            for (int uu = 0; uu < 2; ++uu) {
                float gi = acc[uu*4 + 0], gf_ = acc[uu*4 + 1];
                float gg = acc[uu*4 + 2], go  = acc[uu*4 + 3];
                float cnew = sigf(gf_) * cReg[t*2 + uu] + sigf(gi) * tanha(gg);
                cReg[t*2 + uu] = cnew;
                float hv = sigf(go) * tanha(cnew);
                if (uu == 0) hp0 = (_Float16)hv; else hp1 = (_Float16)hv;
            }
            int u0  = t*4 + hi*2;               // even -> 4B aligned store slot
            int kc2 = u0 >> 5, rem = u0 & 31;
            _Float16* dst = myHT + kc2*512 + ((rem >> 4)*16 + n)*16 + (rem & 15);
            union { _Float16 h2[2]; unsigned u; } pk;
            pk.h2[0] = hp0; pk.h2[1] = hp1;
            *(unsigned*)dst = pk.u;
        }

        // ---- head: logits^T = W_head * h_new^T ; log-softmax over classes ----
        {
            v16h hN[4];
            #pragma unroll
            for (int kc = 0; kc < 4; ++kc) hN[kc] = load16(myHT + kc*512 + L*16);
            const float* hbp = hb + kh*16 + hi*8;
            v8f acc;
            #pragma unroll
            for (int v = 0; v < 8; ++v) acc[v] = hbp[v];
            #pragma unroll
            for (int kc = 0; kc < 4; ++kc) {
                v16h aH = load16(WHD + ((kh*4 + kc)*32 + L)*16);
                acc = __builtin_amdgcn_wmma_f32_16x16x32_f16(false, aH, false, hN[kc],
                                                             (short)0, acc, false, false);
            }
            int nhh  = (kh == 0) ? 15 : ((kh == 1) ? 11 : 10);
            int rowg = blockRow0 + w*16 + n;
            int id   = policy[rowg*30 + outer*3 + kh];

            float l[8], mx = -1.0e30f;
            #pragma unroll
            for (int v = 0; v < 8; ++v) {
                int c = v + 8*hi;
                float lv = (c < nhh) ? TANH_C * tanha(acc[v]) : -1.0e30f;
                l[v] = lv; mx = fmaxf(mx, lv);
            }
            mx = fmaxf(mx, __shfl_xor(mx, 16, 32));
            float se = 0.0f;
            #pragma unroll
            for (int v = 0; v < 8; ++v) se += __expf(l[v] - mx);
            se += __shfl_xor(se, 16, 32);
            float lz = __logf(se);
            float lpc = 0.0f, entc = 0.0f;
            #pragma unroll
            for (int v = 0; v < 8; ++v) {
                int c = v + 8*hi;
                if (c < nhh) {
                    float lg = l[v] - mx - lz;
                    float pp = __expf(lg);
                    entc += pp * lg;
                    if (c == id) lpc += lg;
                }
            }
            lpc  += __shfl_xor(lpc, 16, 32);
            entc += __shfl_xor(entc, 16, 32);
            lpAcc += lpc;
            entAcc -= entc;

            // next x = emb[id]  (lane fills its own B-fragment slot: K = hi*16+e)
            const float* et  = (kh == 0) ? o_emb : ((kh == 1) ? p_emb : m_emb);
            const float* src = et + id*32 + hi*16;
            v8h x0, x1;
            #pragma unroll
            for (int e = 0; e < 8; ++e) { x0[e] = (_Float16)src[e]; x1[e] = (_Float16)src[8+e]; }
            _Float16* xdst = myXT + L*16;
            *(v8h*)xdst       = x0;
            *(v8h*)(xdst + 8) = x1;
        }

        if (++kh == 3) { kh = 0; ++outer; }
    }

    if (hi == 0) {
        int rowg = blockRow0 + w*16 + n;
        out[rowg]           = lpAcc;
        out[B_TOTAL + rowg] = entAcc;
    }
}

__global__ void __launch_bounds__(256)
policy_copy_kernel(const int* __restrict__ pol, float* __restrict__ dst, int nelem)
{
    int idx = blockIdx.x * blockDim.x + threadIdx.x;
    if (idx < nelem) dst[idx] = (float)pol[idx];
}

extern "C" void kernel_launch(void* const* d_in, const int* in_sizes, int n_in,
                              void* d_out, int out_size, void* d_ws, size_t ws_size,
                              hipStream_t stream) {
    (void)in_sizes; (void)n_in; (void)d_ws; (void)ws_size; (void)out_size;
    const int*   policy    = (const int*)  d_in[1];
    const int*   label     = (const int*)  d_in[2];
    const int*   group_fnc = (const int*)  d_in[3];
    const float* in_emb    = (const float*)d_in[4];
    const float* W_ih      = (const float*)d_in[5];
    const float* W_hh      = (const float*)d_in[6];
    const float* b_ih      = (const float*)d_in[7];
    const float* b_hh      = (const float*)d_in[8];
    const float* o_W       = (const float*)d_in[9];
    const float* o_b       = (const float*)d_in[10];
    const float* p_W       = (const float*)d_in[11];
    const float* p_b       = (const float*)d_in[12];
    const float* m_W       = (const float*)d_in[13];
    const float* m_b       = (const float*)d_in[14];
    const float* o_emb     = (const float*)d_in[15];
    const float* p_emb     = (const float*)d_in[16];
    const float* m_emb     = (const float*)d_in[17];
    float* out = (float*)d_out;

    (void)hipFuncSetAttribute((const void*)controller_kernel,
                              hipFuncAttributeMaxDynamicSharedMemorySize, SMEM_BYTES);

    dim3 grid(B_TOTAL / ROWS_BLK), block(128);
    controller_kernel<<<grid, block, SMEM_BYTES, stream>>>(
        policy, label, group_fnc, in_emb, W_ih, W_hh, b_ih, b_hh,
        o_W, o_b, p_W, p_b, m_W, m_b, o_emb, p_emb, m_emb, out);

    int npol = B_TOTAL * 30;
    policy_copy_kernel<<<(npol + 255) / 256, 256, 0, stream>>>(
        policy, out + 2 * B_TOTAL, npol);
}
